// AudioProcessor_28939489641098
// MI455X (gfx1250) — compile-verified, matrix operand order checked by reading, not run on hardware
//
#include <hip/hip_runtime.h>

// ---------------- problem constants ----------------
#define B_ROWS   128
#define T_LEN    262144
#define M_HALF   131072      // complex FFT length (real-pair packing), = 512*256
#define N1       512         // four-step: first-axis FFT length
#define N2       256         // four-step: second-axis FFT length
#define D_MODEL  512
#define L_SM     171         // (512+2-3)//3 + 1
#define PI_F     3.14159265358979323846f

// TDM-padded LDS tile row: 32 float2 data + 1 float2 pad (2 DWORDs per 64 DWORDs)
#define ROW_F2   33

typedef __attribute__((ext_vector_type(2)))  float        v2f;
typedef __attribute__((ext_vector_type(8)))  float        v8f;
typedef __attribute__((ext_vector_type(16))) _Float16     v16h;
typedef __attribute__((ext_vector_type(4)))  unsigned int u32x4;
typedef __attribute__((ext_vector_type(4)))  int          i32x4;
typedef __attribute__((ext_vector_type(8)))  int          i32x8;

#if defined(__AMDGCN__) && __has_builtin(__builtin_amdgcn_tensor_load_to_lds)
#define HAVE_TDM 1
#else
#define HAVE_TDM 0
#endif

#if defined(__AMDGCN__)
#define HAVE_ASYNC_ASM 1
#else
#define HAVE_ASYNC_ASM 0
#endif

__device__ __forceinline__ float2 mkc(float x, float y) { float2 r; r.x = x; r.y = y; return r; }
__device__ __forceinline__ float2 cadd(float2 a, float2 b) { return mkc(a.x + b.x, a.y + b.y); }
__device__ __forceinline__ float2 csub(float2 a, float2 b) { return mkc(a.x - b.x, a.y - b.y); }
__device__ __forceinline__ float2 cmul(float2 a, float2 b) {
  return mkc(a.x * b.x - a.y * b.y, a.x * b.y + a.y * b.x);
}
__device__ __forceinline__ float2 twd(float ang) {
  float s, c; __sincosf(ang, &s, &c); return mkc(c, s);
}
__device__ __forceinline__ unsigned lds_offset(const void* p) {
  // Flat LDS aperture addresses map to LDS via addr[31:0] truncation (ISA 10.2).
  return (unsigned)(size_t)p;
}

// ============================================================
// Kernel 1: four-step stage 1.
// Packed complex z[n], n = n1*256 + n2. For each n2: 512-pt FFT over n1,
// multiply twiddle e^{-2pi i k1 n2 / M}, store transposed T[b][k1][n2].
// Block = 256 threads (8 waves), 32 consecutive n2 columns.
// The 512x32 strided tile (line stride 2KB) is fetched by ONE Tensor-Data-
// Mover descriptor: tile_dim0=32, tile_dim1=512, tensor_dim0_stride=256,
// data_size=8B, pad 2 DWORDs every 64 DWORDs -> LDS row stride 66 DWORDs,
// which makes the butterfly (vary n1) and store (vary c) phases
// bank-conflict-free. In-place DIF radix-2; bit-reversal folded into store.
// ============================================================
__global__ __launch_bounds__(256) void fft_stage1(const float* __restrict__ audio,
                                                  float2* __restrict__ outT) {
  __shared__ float2 s[N1 * ROW_F2];           // elem(n1,c) at s[n1*ROW_F2 + c]
  const int tid  = threadIdx.x;
  const int lane = tid & 31;
  const int wave = tid >> 5;
  const int row  = blockIdx.y;
  const int n2b  = blockIdx.x * 32;

  const float2* z    = (const float2*)(audio) + (size_t)row * M_HALF;
  const float2* gsrc = z + n2b;               // element (n1=0, c=0) of the tile

#if HAVE_TDM
  if (wave == 0) {
    const unsigned           lbase = lds_offset(&s[0]);
    const unsigned long long ga    = (unsigned long long)(size_t)gsrc;
    // ---- D# group 0 (ISA 8.3): count=1 | lds_addr | global_addr | type=2 ----
    u32x4 g0;
    g0[0] = 1u;                                              // count=1, user mode
    g0[1] = lbase;                                           // lds_addr (bytes)
    g0[2] = (unsigned)(ga & 0xffffffffu);                    // global_addr[31:0]
    g0[3] = (unsigned)((ga >> 32) & 0x01ffffffu) | (2u << 30); // ga[56:32] | type=2
    // ---- D# group 1 (ISA 8.4) ----
    i32x8 g1;
    g1[0] = (int)((3u << 16)      // data_size = 8 bytes
                | (1u << 20)      // pad_enable
                | (5u << 22)      // pad_interval: every 64 DWORDs (=one tile row)
                | (1u << 25));    // pad_amount: 2 DWORDs
    g1[1] = (int)(256u << 16);                       // tensor_dim0 = 256 (lo16)
    g1[2] = (int)(512u << 16);                       // dim0 hi16=0 | tensor_dim1=512 lo16
    g1[3] = (int)(32u << 16);                        // dim1 hi16=0 | tile_dim0 = 32
    g1[4] = (int)512u;                               // tile_dim1 = 512, tile_dim2 = 0
    g1[5] = (int)256;                                // tensor_dim0_stride = 256 (lo32)
    g1[6] = 0;                                       // stride hi16 | dim1_stride lo16
    g1[7] = 0;                                       // dim1_stride hi32
    const i32x4 z4 = {0, 0, 0, 0};
#if __clang_major__ >= 23
    const i32x8 z8 = {0, 0, 0, 0, 0, 0, 0, 0};
    __builtin_amdgcn_tensor_load_to_lds(g0, g1, z4, z4, z8, 0);
#else
    __builtin_amdgcn_tensor_load_to_lds(g0, g1, z4, z4, 0);
#endif
    __builtin_amdgcn_s_wait_tensorcnt(0);            // s_wait_tensorcnt 0
  }
  __syncthreads();
#else
  // Fallback (host pass / no TDM): cooperative coalesced load into same layout.
  for (int it = 0; it < (32 * N1) / 256; ++it) {
    int idx = tid + 256 * it;
    int c   = idx & 31;
    int n1  = idx >> 5;
    s[n1 * ROW_F2 + c] = z[(size_t)n1 * N2 + (n2b + c)];
  }
  __syncthreads();
#endif

  // 9 in-place DIF radix-2 stages; each wave owns 4 columns.
  for (int h = N1 >> 1; h >= 1; h >>= 1) {
    for (int cc = 0; cc < 4; ++cc) {
      int c = wave * 4 + cc;
      for (int t = lane; t < N1 / 2; t += 32) {
        int k = t & (h - 1);
        int i = ((t - k) << 1) + k;          // g*2h + k
        float2 a  = s[i * ROW_F2 + c];
        float2 bz = s[(i + h) * ROW_F2 + c];
        float2 w  = twd(-PI_F * (float)k / (float)h);
        s[i * ROW_F2 + c]       = cadd(a, bz);
        s[(i + h) * ROW_F2 + c] = cmul(csub(a, bz), w);
      }
    }
    __syncthreads();
  }

  // Twiddle + transpose store; natural k1 read from bit-reversed LDS slot.
  const float wM = -2.0f * PI_F / (float)M_HALF;
  for (int it = 0; it < (32 * N1) / 256; ++it) {
    int idx = tid + 256 * it;
    int c   = idx & 31;
    int k1  = idx >> 5;
    int p   = (int)(__brev((unsigned)k1) >> 23);   // 9-bit reverse
    float2 y  = s[p * ROW_F2 + c];
    int    n2 = n2b + c;
    // k1*n2 < 512*256 = M, so |angle| <= 2pi: no range-reduction trouble.
    y = cmul(y, twd(wM * (float)(k1 * n2)));
    outT[((size_t)row * N1 + k1) * N2 + n2] = y;   // coalesced (lanes -> n2)
  }
}

// ============================================================
// Kernel 2: four-step stage 2.
// 256-pt FFT over each contiguous row T[b][k1][.]; keep [k1][k2] layout
// (bin k = k1 + 512*k2 lives at offset k1*256 + k2).
// Block = 256 threads, 8 waves, one k1-row per wave. LDS fill uses per-lane
// async global->LDS copies (ASYNCcnt) instead of load-to-VGPR-store-to-LDS.
// ============================================================
__global__ __launch_bounds__(256) void fft_stage2(const float2* __restrict__ inT,
                                                  float2* __restrict__ outZ) {
  __shared__ float2 s[8][N2 + 1];
  const int lane = threadIdx.x & 31;
  const int wave = threadIdx.x >> 5;
  const int row  = blockIdx.y;
  const int k1   = blockIdx.x * 8 + wave;

  const float2* src = inT + ((size_t)row * N1 + k1) * N2;

#if HAVE_ASYNC_ASM
  {
    const unsigned lbase = lds_offset(&s[wave][0]);
    for (int j = 0; j < N2 / 32; ++j) {
      int idx = lane + 32 * j;
      unsigned           loff = lbase + (unsigned)idx * 8u;
      unsigned long long ga   = (unsigned long long)(size_t)(src + idx);
      asm volatile("global_load_async_to_lds_b64 %0, %1, off"
                   :: "v"(loff), "v"(ga) : "memory");
    }
    asm volatile("s_wait_asynccnt 0x0" ::: "memory");
  }
  __syncthreads();
#else
  for (int j = 0; j < N2 / 32; ++j) s[wave][lane + 32 * j] = src[lane + 32 * j];
  __syncthreads();
#endif

  for (int h = N2 >> 1; h >= 1; h >>= 1) {
    for (int t = lane; t < N2 / 2; t += 32) {
      int k = t & (h - 1);
      int i = ((t - k) << 1) + k;
      float2 a  = s[wave][i];
      float2 bz = s[wave][i + h];
      float2 w  = twd(-PI_F * (float)k / (float)h);
      s[wave][i]     = cadd(a, bz);
      s[wave][i + h] = cmul(csub(a, bz), w);
    }
    __syncthreads();
  }

  // Natural-order contiguous store (gather bit-reversed from LDS).
  float2* Zr = outZ + ((size_t)row * N1 + k1) * N2;
  for (int j = 0; j < N2 / 32; ++j) {
    int k2 = lane + 32 * j;
    int p  = (int)(__brev((unsigned)k2) >> 24);    // 8-bit reverse
    Zr[k2] = s[wave][p];
  }
}

// ============================================================
// Kernel 3: spectral post-processing, one block (512 threads) per row.
// Gathers only the ~1024 FFT bins the 512-wide interp needs, applies the
// real-FFT unpack, then smooth/peak/interp entirely in LDS.
// ============================================================
__device__ __forceinline__ float2 zread(const float2* Zc, int k /* 0..M-1 */) {
  return Zc[(size_t)(k & (N1 - 1)) * N2 + (k >> 9)];
}

__device__ __forceinline__ float rfft_mag(const float2* Zc, int k /* 0..M */) {
  int ka = k & (M_HALF - 1);
  int kb = (M_HALF - k) & (M_HALF - 1);
  float2 zk = zread(Zc, ka);
  float2 zm = zread(Zc, kb);
  float2 e = mkc(0.5f * (zk.x + zm.x), 0.5f * (zk.y - zm.y));
  float2 d = mkc(zk.x - zm.x, zk.y + zm.y);
  float2 o = mkc(0.5f * d.y, -0.5f * d.x);           // d / (2i)
  float2 x = cadd(e, cmul(twd(-PI_F * (float)k / (float)M_HALF), o));
  return sqrtf(x.x * x.x + x.y * x.y);
}

__global__ __launch_bounds__(512) void spectral_post(const float2* __restrict__ Zall,
                                                     float* __restrict__ peaks_out) {
  __shared__ float spec[D_MODEL];
  __shared__ float smoothed[L_SM];
  __shared__ float pk[L_SM];
  const int row = blockIdx.x;
  const int t   = threadIdx.x;
  const float2* Z = Zall + (size_t)row * M_HALF;

  { // linear_interp(mag, 512), in_size = M+1 = 131073
    const float scale = (float)(M_HALF + 1) / (float)D_MODEL;
    float src = ((float)t + 0.5f) * scale - 0.5f;
    src = fmaxf(src, 0.0f);
    int i0 = min((int)floorf(src), M_HALF);
    int i1 = min(i0 + 1, M_HALF);
    float w = src - (float)i0;
    spec[t] = rfft_mag(Z, i0) * (1.0f - w) + rfft_mag(Z, i1) * w;
  }
  __syncthreads();

  if (t < L_SM) { // padded[:513].reshape(171,3).mean
    float acc = 0.0f;
#pragma unroll
    for (int u = 0; u < 3; ++u) {
      int i = 3 * t + u;                       // padded index
      acc += (i >= 1 && i <= D_MODEL) ? spec[i - 1] : 0.0f;
    }
    smoothed[t] = acc * (1.0f / 3.0f);
  }
  __syncthreads();

  if (t < L_SM) { // strict local maxima, zero-padded ends
    float v = 0.0f;
    if (t >= 1 && t <= L_SM - 2) {
      float c = smoothed[t];
      if (c > smoothed[t - 1] && c > smoothed[t + 1]) v = c;
    }
    pk[t] = v;
  }
  __syncthreads();

  { // linear_interp(peaks, 512), in_size = 171
    const float scale = (float)L_SM / (float)D_MODEL;
    float src = ((float)t + 0.5f) * scale - 0.5f;
    src = fmaxf(src, 0.0f);
    int i0 = min((int)floorf(src), L_SM - 1);
    int i1 = min(i0 + 1, L_SM - 1);
    float w = src - (float)i0;
    peaks_out[(size_t)row * D_MODEL + t] = pk[i0] * (1.0f - w) + pk[i1] * w;
  }
}

// ============================================================
// Kernel 4: out = peaks @ W^T + b via fp32 WMMA (V_WMMA_F32_16X16X4_F32).
// 8x32 = 256 16x16 tiles, one wave per tile (32 blocks x 8 waves).
// K=512 chained through 128 accumulating WMMAs; B-rows prefetched.
// ============================================================
__global__ __launch_bounds__(256) void gemm_wmma(const float* __restrict__ P,
                                                 const float* __restrict__ W,
                                                 const float* __restrict__ bias,
                                                 float* __restrict__ out) {
  const int lane = threadIdx.x & 31;
  const int wave = threadIdx.x >> 5;
  const int gid  = blockIdx.x * 8 + wave;   // tile id 0..255
  const int tm   = gid >> 5;                // 0..7   (M tiles)
  const int tn   = gid & 31;                // 0..31  (N tiles)
  const int mrow = tm * 16 + (lane & 15);
  const int ncol = tn * 16 + (lane & 15);

  v8f acc = {};

#if __has_builtin(__builtin_amdgcn_wmma_f32_16x16x4_f32)
  // f32 16x16x4: A lanes 0-15 hold K={0,1}, lanes 16-31 hold K={2,3} (M = lane&15).
  const int kof = (lane >> 4) << 1;                 // 0 or 2
  const float* Ar = P + (size_t)mrow * D_MODEL + kof;
  const float* Br = W + (size_t)ncol * D_MODEL + kof;   // B[k][n] = W[n][k]
  for (int kk = 0; kk < D_MODEL / 4; ++kk) {
    __builtin_prefetch(Br + kk * 4 + 64, 0, 1);     // global_prefetch_b8
    v2f a = { Ar[kk * 4 + 0], Ar[kk * 4 + 1] };
    v2f b = { Br[kk * 4 + 0], Br[kk * 4 + 1] };
    acc = __builtin_amdgcn_wmma_f32_16x16x4_f32(false, a, false, b,
                                                (short)0, acc, false, false);
  }
#else
  // f16 16x16x32 fallback: lanes 0-15 hold K 0-7 & 16-23, lanes 16-31 K 8-15 & 24-31.
  const int khalf = (lane >> 4) << 3;               // 0 or 8
  for (int kk = 0; kk < D_MODEL / 32; ++kk) {
    int base = kk * 32 + khalf;
    v16h av, bv;
#pragma unroll
    for (int e = 0; e < 8; ++e) {
      av[e]     = (_Float16)P[(size_t)mrow * D_MODEL + base + e];
      av[8 + e] = (_Float16)P[(size_t)mrow * D_MODEL + base + 16 + e];
      bv[e]     = (_Float16)W[(size_t)ncol * D_MODEL + base + e];
      bv[8 + e] = (_Float16)W[(size_t)ncol * D_MODEL + base + 16 + e];
    }
    acc = __builtin_amdgcn_wmma_f32_16x16x32_f16(false, av, false, bv,
                                                 (short)0, acc, false, false);
  }
#endif

  // C/D layout: VGPR r -> M = r (lanes 0-15) or r+8 (lanes 16-31), N = lane&15.
  const float bv0   = bias[ncol];
  const int   mbase = tm * 16 + ((lane >> 4) << 3);
#pragma unroll
  for (int r = 0; r < 8; ++r)
    out[(size_t)(mbase + r) * D_MODEL + ncol] = acc[r] + bv0;
}

// ============================================================
// launch
// ============================================================
extern "C" void kernel_launch(void* const* d_in, const int* in_sizes, int n_in,
                              void* d_out, int out_size, void* d_ws, size_t ws_size,
                              hipStream_t stream) {
  const float* audio = (const float*)d_in[0];   // (128, 262144) f32
  const float* W     = (const float*)d_in[1];   // (512, 512)    f32
  const float* bias  = (const float*)d_in[2];   // (512,)        f32
  float*       out   = (float*)d_out;           // (128, 512)    f32

  // Workspace: two 134 MB complex buffers (ping/pong for the two FFT passes).
  float2* bufA  = (float2*)d_ws;                      // T matrix [B][512][256]
  float2* bufB  = bufA + (size_t)B_ROWS * M_HALF;     // Z (k1-major) [B][512][256]
  float*  peaks = (float*)d_ws;                       // 256 KB, reuses bufA (dead after stage2)

  fft_stage1   <<<dim3(N2 / 32, B_ROWS), 256, 0, stream>>>(audio, bufA);
  fft_stage2   <<<dim3(N1 / 8,  B_ROWS), 256, 0, stream>>>(bufA, bufB);
  spectral_post<<<dim3(B_ROWS),          512, 0, stream>>>(bufB, peaks);
  gemm_wmma    <<<dim3(32),              256, 0, stream>>>(peaks, W, bias, out);
}
